// IntermediateBlock2_80994493268106
// MI455X (gfx1250) — matrix operand order repaired and checked by reading, not compiled, not run on hardware
//
#include <hip/hip_runtime.h>
#include <hip/hip_bf16.h>

// ---------------------------------------------------------------------------
// MI455X (gfx1250) implementation: implicit-GEMM 3x3 convs on
// v_wmma_f32_16x16x32_bf16 (bf16 in, f32 accum), plus small VALU kernels for
// attention softmax, ROI-align and the final combine.
// ---------------------------------------------------------------------------

typedef __attribute__((ext_vector_type(16))) __bf16 v16bf;
typedef __attribute__((ext_vector_type(8)))  __bf16 v8bf;
typedef __attribute__((ext_vector_type(8)))  float  v8f;

#define HWDIM 64      // input spatial size (all conv inputs are 64x64)
#define BATCH 32
#define OCH   128

__device__ __forceinline__ unsigned f2bf_rne_u(float f) {
    union { float f; unsigned u; } v; v.f = f;
    unsigned u = v.u;
    u += 0x7FFFu + ((u >> 16) & 1u);   // round to nearest even
    return u >> 16;
}

// rem in [0,9) -> kh = rem/3, kw = rem%3 via packed 2-bit LUTs (no divides)
__device__ __forceinline__ int kh_of(int rem) { return (0x2A540 >> (rem << 1)) & 3; }
__device__ __forceinline__ int kw_of(int rem) { return (0x24924 >> (rem << 1)) & 3; }

// ---------------------------------------------------------------------------
// fp32 -> bf16 weight pack (flat OIHW -> [OC][IC*9] bf16, same flat order)
// ---------------------------------------------------------------------------
__global__ void pack_bf16(const float* __restrict__ src,
                          unsigned short* __restrict__ dst, int n) {
    int i = blockIdx.x * blockDim.x + threadIdx.x;
    if (i < n) dst[i] = (unsigned short)f2bf_rne_u(src[i]);
}

// ---------------------------------------------------------------------------
// Implicit-GEMM 3x3 conv, pad=1, STRIDE in {1,2}, + bias + leaky-relu(0.1).
//   M = OC = 128, N = BATCH*Ho*Wo, K = (Ca+Cb)*9
// Workgroup: 128 threads (4 waves). WG tile 128(M) x 64(N); each wave owns
// 128(M) x 16(N) via eight 16x16 f32 accumulators, so one staged B tile
// feeds 32 WMMAs. CONCAT=true reads channels [0,Ca) from xa and [Ca,Ca+Cb)
// from xb (channel-concat convs without materializing the concat).
// ---------------------------------------------------------------------------
template <bool CONCAT, int STRIDE, int LW>
__global__ __launch_bounds__(128)
void conv3x3_lrelu_wmma(const float* __restrict__ xa,
                        const float* __restrict__ xb,
                        int Ca, int Cb,
                        const unsigned short* __restrict__ wpk, // bf16 [OC][Ktot]
                        const float* __restrict__ bias,
                        float* __restrict__ out)
{
    const int Ktot = (CONCAT ? (Ca + Cb) : Ca) * 9;
    constexpr int Wo = 1 << LW;
    constexpr int Ho = Wo;

    // B tile: [n=64][k=32] bf16, stored as packed dword pairs [n][16]
    __shared__ __align__(16) unsigned btile32[64 * 16];

    const int t     = threadIdx.x;
    const int lane  = t & 31;
    const int wave  = t >> 5;
    const int pBase = blockIdx.x * 64;   // N tile base (pixel index)

    v8f acc[8];
    #pragma unroll
    for (int s = 0; s < 8; ++s) acc[s] = v8f{};

    // ---- staging thread -> (n, k-pair) assignment ----
    // thread t stages pixel n = t&63, k-pairs pp = pair0 + 2*it (it = 0..7),
    // i.e. k = 2*pair0 + 4*it within each 32-K tile; the global k sequence
    // advances by exactly 4 per iteration, seamlessly across K tiles, so the
    // (ic, rem = kh*3+kw) decomposition is carried incrementally (no divides).
    const int nStage = t & 63;
    const int pair0  = t >> 6;          // 0 or 1
    const int pS     = pBase + nStage;
    const int woS    = pS & (Wo - 1);
    const int hoS    = (pS >> LW) & (Ho - 1);
    const int bS     = pS >> (2 * LW);
    const int hiBase = hoS * STRIDE - 1;
    const int wiBase = woS * STRIDE - 1;

    int icS  = 0;
    int remS = 2 * pair0;               // 0 or 2 (< 9)

    // fragment geometry (constant across K loop)
    const int nSub  = wave * 16 + (lane & 15);
    const int koff  = (lane >> 4) * 16;
    const int mRow  = lane & 15;
    const int khalf = (lane >> 4) * 8;

    for (int k0 = 0; k0 < Ktot; k0 += 32) {
        // ---- stage B tile: 64 pixels x 32 K (bf16), zero-padded borders ----
        #pragma unroll
        for (int it = 0; it < 8; ++it) {
            const int ic0 = icS, rem0 = remS;
            int ic1 = ic0, rem1 = rem0 + 1;
            if (rem1 == 9) { rem1 = 0; ++ic1; }
            remS += 4; if (remS >= 9) { remS -= 9; ++icS; }

            float v0 = 0.0f, v1 = 0.0f;
            {
                int hi = hiBase + kh_of(rem0), wi = wiBase + kw_of(rem0);
                if ((unsigned)hi < (unsigned)HWDIM && (unsigned)wi < (unsigned)HWDIM) {
                    if (CONCAT && ic0 >= Ca)
                        v0 = xb[(((size_t)bS * Cb + (ic0 - Ca)) * HWDIM + hi) * HWDIM + wi];
                    else
                        v0 = xa[(((size_t)bS * Ca + ic0) * HWDIM + hi) * HWDIM + wi];
                }
            }
            {
                int hi = hiBase + kh_of(rem1), wi = wiBase + kw_of(rem1);
                if ((unsigned)hi < (unsigned)HWDIM && (unsigned)wi < (unsigned)HWDIM) {
                    if (CONCAT && ic1 >= Ca)
                        v1 = xb[(((size_t)bS * Cb + (ic1 - Ca)) * HWDIM + hi) * HWDIM + wi];
                    else
                        v1 = xa[(((size_t)bS * Ca + ic1) * HWDIM + hi) * HWDIM + wi];
                }
            }
            btile32[nStage * 16 + pair0 + it * 2] =
                f2bf_rne_u(v0) | (f2bf_rne_u(v1) << 16);
        }
        __syncthreads();

        // ---- B fragment (KxN 32x16, same K order as A): lane n = lane&15,
        //      contiguous K run [koff, koff+16) with koff = (lane>=16)*16 ----
        const v8bf* bp =
            (const v8bf*)((const unsigned short*)btile32 + nSub * 32 + koff);
        v8bf b0 = bp[0], b1 = bp[1];
        v16bf bf = __builtin_shufflevector(b0, b1,
            0, 1, 2, 3, 4, 5, 6, 7, 8, 9, 10, 11, 12, 13, 14, 15);

        // ---- preload all eight A fragments (MxK 16x32): lane m = lane&15,
        //      contiguous K run [koff, koff+16) ----
        v16bf af[8];
        #pragma unroll
        for (int sub = 0; sub < 8; ++sub) {
            const unsigned short* wrow =
                wpk + (size_t)(sub * 16 + mRow) * Ktot + k0;
            v8bf a0 = *(const v8bf*)(wrow + koff);
            v8bf a1 = *(const v8bf*)(wrow + koff + 8);
            af[sub] = __builtin_shufflevector(a0, a1,
                0, 1, 2, 3, 4, 5, 6, 7, 8, 9, 10, 11, 12, 13, 14, 15);
        }
        // prefetch next K-step of the weight rows (global_prefetch_b8 hint)
        if (k0 + 32 < Ktot)
            __builtin_prefetch(wpk + (size_t)mRow * Ktot + k0 + 32, 0, 1);

        // ---- 8 back-to-back WMMAs sharing the B fragment ----
        #pragma unroll
        for (int sub = 0; sub < 8; ++sub)
            acc[sub] = __builtin_amdgcn_wmma_f32_16x16x32_bf16(
                false, af[sub], false, bf, (short)0, acc[sub], false, false);

        __syncthreads();
    }

    // ---- epilogue: bias + leaky-relu, NCHW fp32 store ----
    // C/D layout: lanes 0-15 -> N=lane, M=r ; lanes 16-31 -> N=lane-16, M=8+r
    const int nOut = pBase + wave * 16 + (lane & 15);
    const int woO  = nOut & (Wo - 1);
    const int hoO  = (nOut >> LW) & (Ho - 1);
    const int bO   = nOut >> (2 * LW);
    const int mHi  = (lane >> 4) * 8;
    #pragma unroll
    for (int sub = 0; sub < 8; ++sub) {
        const v8f bv = *(const v8f*)(bias + sub * 16 + mHi);   // 8 contiguous biases
        #pragma unroll
        for (int r = 0; r < 8; ++r) {
            int oc  = sub * 16 + mHi + r;
            float v = acc[sub][r] + bv[r];
            v = (v >= 0.0f) ? v : 0.1f * v;
            out[(((size_t)bO * OCH + oc) * Ho + hoO) * Wo + woO] = v;
        }
    }
}

// ---------------------------------------------------------------------------
// Attention logits: logits[b,p] = <lin_w[p,:], out1[b,:,p]> + lin_b[p]
// ---------------------------------------------------------------------------
__global__ void attn_logits(const float* __restrict__ out1,
                            const float* __restrict__ lin_w,
                            const float* __restrict__ lin_b,
                            float* __restrict__ logits)
{
    int p = blockIdx.x * blockDim.x + threadIdx.x;   // 0..4095
    int b = blockIdx.y;
    const float* xp = out1 + (size_t)b * 64 * 4096 + p;
    const float* wp = lin_w + (size_t)p * 64;
    float s = lin_b[p];
    #pragma unroll 8
    for (int c = 0; c < 64; ++c) s += xp[(size_t)c * 4096] * wp[c];
    logits[(size_t)b * 4096 + p] = s;
}

// ---------------------------------------------------------------------------
// Softmax over 4096 positions, one block per batch element.
// ---------------------------------------------------------------------------
__global__ __launch_bounds__(256)
void softmax4096(const float* __restrict__ logits, float* __restrict__ attn)
{
    __shared__ float red[256];
    const int b = blockIdx.x;
    const float* lp = logits + (size_t)b * 4096;

    float m = -3.4e38f;
    for (int p = threadIdx.x; p < 4096; p += 256) m = fmaxf(m, lp[p]);
    red[threadIdx.x] = m; __syncthreads();
    for (int s = 128; s > 0; s >>= 1) {
        if (threadIdx.x < s) red[threadIdx.x] = fmaxf(red[threadIdx.x], red[threadIdx.x + s]);
        __syncthreads();
    }
    m = red[0]; __syncthreads();

    float sum = 0.0f;
    for (int p = threadIdx.x; p < 4096; p += 256) sum += __expf(lp[p] - m);
    red[threadIdx.x] = sum; __syncthreads();
    for (int s = 128; s > 0; s >>= 1) {
        if (threadIdx.x < s) red[threadIdx.x] += red[threadIdx.x + s];
        __syncthreads();
    }
    const float inv = 1.0f / red[0];
    for (int p = threadIdx.x; p < 4096; p += 256)
        attn[(size_t)b * 4096 + p] = __expf(lp[p] - m) * inv;
}

// ---------------------------------------------------------------------------
// ROI-align helpers (64x64 plane, output 32x32, sampling_ratio=2)
// ---------------------------------------------------------------------------
__device__ __forceinline__ float roi_bilin(const float* __restrict__ base,
                                           float y, float x) {
    int y0 = (int)floorf(y), x0 = (int)floorf(x);
    int y1 = min(y0 + 1, 63), x1 = min(x0 + 1, 63);
    float ly = y - (float)y0, lx = x - (float)x0;
    float hy = 1.0f - ly, hx = 1.0f - lx;
    return hy * hx * base[y0 * 64 + x0] + hy * lx * base[y0 * 64 + x1]
         + ly * hx * base[y1 * 64 + x0] + ly * lx * base[y1 * 64 + x1];
}

__device__ __forceinline__ float roi_bilin_attn(const float* __restrict__ base,
                                                const float* __restrict__ ap,
                                                float y, float x) {
    int y0 = (int)floorf(y), x0 = (int)floorf(x);
    int y1 = min(y0 + 1, 63), x1 = min(x0 + 1, 63);
    float ly = y - (float)y0, lx = x - (float)x0;
    float hy = 1.0f - ly, hx = 1.0f - lx;
    int i00 = y0 * 64 + x0, i01 = y0 * 64 + x1, i10 = y1 * 64 + x0, i11 = y1 * 64 + x1;
    return hy * hx * base[i00] * ap[i00] + hy * lx * base[i01] * ap[i01]
         + ly * hx * base[i10] * ap[i10] + ly * lx * base[i11] * ap[i11];
}

// ---------------------------------------------------------------------------
// Final combine: bo1 = roi(l2) + aux ; bo2 = b2 ; bo3 = roi(prev*attn)
// out = stack([bo1+bo2+bo3, bo2, bo3])   shape [3][B][128][32][32]
// ---------------------------------------------------------------------------
__global__ __launch_bounds__(256)
void combine_kernel(const float* __restrict__ l2,
                    const float* __restrict__ aux,
                    const float* __restrict__ b2,
                    const float* __restrict__ prev,
                    const float* __restrict__ attn,
                    float* __restrict__ out)
{
    int idx = blockIdx.x * blockDim.x + threadIdx.x;   // over 32*128*32*32
    int ow = idx & 31;
    int oh = (idx >> 5) & 31;
    int oc = (idx >> 10) & 127;
    int b  = idx >> 17;

    const float binf = 63.0f / 32.0f;   // bin size (h-1)/ph
    const float* l2p = l2   + ((size_t)(b * OCH + oc) << 12);
    const float* pvp = prev + ((size_t)(b * OCH + oc) << 12);
    const float* ap  = attn + ((size_t)b << 12);

    float r1 = 0.0f, r3 = 0.0f;
    #pragma unroll
    for (int sy = 0; sy < 2; ++sy) {
        #pragma unroll
        for (int sx = 0; sx < 2; ++sx) {
            float y = ((float)oh + ((float)sy + 0.5f) * 0.5f) * binf;
            float x = ((float)ow + ((float)sx + 0.5f) * 0.5f) * binf;
            r1 += roi_bilin(l2p, y, x);
            r3 += roi_bilin_attn(pvp, ap, y, x);
        }
    }
    r1 *= 0.25f; r3 *= 0.25f;

    float bo1 = r1 + aux[idx];
    float bo2 = b2[idx];
    const size_t S = (size_t)BATCH * OCH * 32 * 32;
    out[idx]         = bo1 + bo2 + r3;
    out[S + idx]     = bo2;
    out[2 * S + idx] = r3;
}

// ---------------------------------------------------------------------------
// Host launcher
// ---------------------------------------------------------------------------
extern "C" void kernel_launch(void* const* d_in, const int* in_sizes, int n_in,
                              void* d_out, int out_size, void* d_ws, size_t ws_size,
                              hipStream_t stream)
{
    (void)in_sizes; (void)n_in; (void)out_size; (void)ws_size;

    const float* out1   = (const float*)d_in[0];
    const float* out2   = (const float*)d_in[1];
    const float* out3   = (const float*)d_in[2];
    const float* w1     = (const float*)d_in[3];
    const float* b1     = (const float*)d_in[4];
    const float* w2     = (const float*)d_in[5];
    const float* b2w    = (const float*)d_in[6];
    const float* w3     = (const float*)d_in[7];
    const float* b3     = (const float*)d_in[8];
    const float* w_aux  = (const float*)d_in[9];
    const float* b_aux  = (const float*)d_in[10];
    const float* w_b2   = (const float*)d_in[11];
    const float* b_b2   = (const float*)d_in[12];
    const float* w_prev = (const float*)d_in[13];
    const float* b_prev = (const float*)d_in[14];
    const float* lin_w  = (const float*)d_in[15];
    const float* lin_b  = (const float*)d_in[16];

    const size_t NFULL = (size_t)BATCH * OCH * 64 * 64;   // 16,777,216 floats
    const size_t NHALF = (size_t)BATCH * OCH * 32 * 32;   //  4,194,304 floats

    float* l1     = (float*)d_ws;
    float* l2v    = l1   + NFULL;
    float* l3v    = l2v  + NFULL;      // later reused for `prev`
    float* auxb   = l3v  + NFULL;
    float* b2b    = auxb + NHALF;
    float* logits = b2b  + NHALF;
    float* attn   = logits + (size_t)BATCH * 4096;

    unsigned short* w1p  = (unsigned short*)(attn + (size_t)BATCH * 4096);
    unsigned short* w2p  = w1p  + 128 * 576;
    unsigned short* w3p  = w2p  + 128 * 1152;
    unsigned short* waxp = w3p  + 128 * 1152;
    unsigned short* wb2p = waxp + 128 * 1152;
    unsigned short* wpvp = wb2p + 128 * 2304;

    // --- pack weights to bf16 ---
    {
        struct { const float* s; unsigned short* d; int n; } jobs[6] = {
            { w1,     w1p,  128 * 576  },
            { w2,     w2p,  128 * 1152 },
            { w3,     w3p,  128 * 1152 },
            { w_aux,  waxp, 128 * 1152 },
            { w_b2,   wb2p, 128 * 2304 },
            { w_prev, wpvp, 128 * 1152 },
        };
        for (int j = 0; j < 6; ++j)
            pack_bf16<<<dim3((jobs[j].n + 255) / 256), dim3(256), 0, stream>>>(
                jobs[j].s, jobs[j].d, jobs[j].n);
    }

    const dim3 cblk(128);
    const dim3 gFull(2048);   // N = 32*4096 -> 2048 tiles of 64 pixels
    const dim3 gHalf(512);    // N = 32*1024 ->  512 tiles of 64 pixels

    // l1 = lrelu(conv(out1, w1))
    conv3x3_lrelu_wmma<false, 1, 6><<<gFull, cblk, 0, stream>>>(
        out1, out1, 64, 0, w1p, b1, l1);
    // l2 = lrelu(conv(l1, w2))
    conv3x3_lrelu_wmma<false, 1, 6><<<gFull, cblk, 0, stream>>>(
        l1, l1, 128, 0, w2p, b2w, l2v);
    // l3 = lrelu(conv(l2, w3))
    conv3x3_lrelu_wmma<false, 1, 6><<<gFull, cblk, 0, stream>>>(
        l2v, l2v, 128, 0, w3p, b3, l3v);
    // aux = lrelu(conv(l3, w_aux, stride 2))
    conv3x3_lrelu_wmma<false, 2, 5><<<gHalf, cblk, 0, stream>>>(
        l3v, l3v, 128, 0, waxp, b_aux, auxb);
    // b2 = lrelu(conv(concat(l1,l2), w_b2, stride 2))
    conv3x3_lrelu_wmma<true, 2, 5><<<gHalf, cblk, 0, stream>>>(
        l1, l2v, 128, 128, wb2p, b_b2, b2b);
    // prev = lrelu(conv(concat(out2,out3), w_prev))  (overwrites l3 region)
    conv3x3_lrelu_wmma<true, 1, 6><<<gFull, cblk, 0, stream>>>(
        out2, out3, 64, 64, wpvp, b_prev, l3v);

    // attention
    attn_logits<<<dim3(16, 32), dim3(256), 0, stream>>>(out1, lin_w, lin_b, logits);
    softmax4096<<<dim3(32), dim3(256), 0, stream>>>(logits, attn);

    // combine + both ROI-aligns + stacked output
    combine_kernel<<<dim3((unsigned)(NHALF / 256)), dim3(256), 0, stream>>>(
        l2v, auxb, b2b, l3v, attn, (float*)d_out);
}
